// GraphConvolutionNetwork_85693187490256
// MI455X (gfx1250) — matrix operand order repaired and checked by reading, not compiled, not run on hardware
//
#include <hip/hip_runtime.h>

typedef _Float16 v16h __attribute__((ext_vector_type(16)));
typedef _Float16 v8h  __attribute__((ext_vector_type(8)));
typedef float    v8f  __attribute__((ext_vector_type(8)));
typedef float    v2f  __attribute__((ext_vector_type(2)));

#define N4K 4096
#define DD  128
#define NCOL 256

// LDS staging geometry for GEMM1 B tiles: 64 cols x 64B of K-data per matrix,
// padded to 80B/col (20 dwords -> conflict-free across 64 banks).
#define COL_STRIDE 80
#define MAT_BYTES  (64 * COL_STRIDE)   // 5120
#define BUF_BYTES  (2 * MAT_BYTES)     // 10240 (Xh | Xl)
#define LDS_BYTES  (2 * BUF_BYTES)     // 20480 double-buffered

// ---------------- zero the BN accumulators (256 floats: sum | sumsq) -------
__global__ void k_zero(float* p) { p[threadIdx.x] = 0.0f; }

// ---------------- binarize A -> f16 (exact), row sums -> dinv --------------
__global__ void k_binarize(const float* __restrict__ A, _Float16* __restrict__ Ah,
                           float* __restrict__ dinv) {
    const int lane = threadIdx.x & 31;
    const int wave = threadIdx.x >> 5;
    const int row  = blockIdx.x * 8 + wave;
    const float*  ap = A  + (size_t)row * N4K;
    _Float16*     hp = Ah + (size_t)row * N4K;
    float s = 0.0f;
    for (int c = lane * 2; c < N4K; c += 64) {
        float a0 = ap[c], a1 = ap[c + 1];
        float b0 = a0 > 0.0f ? 1.0f : 0.0f;
        float b1 = a1 > 0.0f ? 1.0f : 0.0f;
        s += b0 + b1;
        union { _Float16 h[2]; unsigned u; } pk;
        pk.h[0] = (_Float16)b0; pk.h[1] = (_Float16)b1;
        *(unsigned*)(hp + c) = pk.u;
    }
    for (int off = 16; off > 0; off >>= 1) s += __shfl_xor(s, off, 32);
    if (lane == 0) dinv[row] = rsqrtf(s + 1.0f);   // deg = rowsum(A)+1
}

// ---------------- build Xs^T (N-major), split f16 hi/lo --------------------
__global__ void k_build_xs(const float* __restrict__ x, const float* __restrict__ dinv,
                           _Float16* __restrict__ Xh, _Float16* __restrict__ Xl) {
    int idx = blockIdx.x * blockDim.x + threadIdx.x;   // == nc*4096 + k
    int k   = idx & (N4K - 1);
    int nc  = idx >> 12;
    int srcRow = (nc < DD) ? k : (N4K + k);
    float v = dinv[k] * x[(size_t)srcRow * DD + (nc & (DD - 1))];
    _Float16 hi = (_Float16)v;
    Xh[idx] = hi;
    Xl[idx] = (_Float16)(v - (float)hi);
}

// ---------------- GEMM1: S[4096,256] = Ah @ (Xs_hi + Xs_lo) ----------------
// block = 8 waves; wave computes 16(M) x 64(N); grid = (4096/128, 256/64).
// B tiles staged cooperatively into LDS with async global->LDS copies,
// double-buffered against the WMMA consumption of the previous step.
__global__ void __launch_bounds__(256) k_gemm1(const _Float16* __restrict__ Ah,
                                               const _Float16* __restrict__ Xh,
                                               const _Float16* __restrict__ Xl,
                                               float* __restrict__ S) {
    __shared__ __align__(16) unsigned char smem[LDS_BYTES];

    const int tid  = threadIdx.x;
    const int lane = tid & 31;
    const int wave = tid >> 5;
    const int r  = lane & 15;          // row/col within tile
    const int hh = lane >> 4;          // lane half selects K sub-block
    const int m0 = blockIdx.x * 128 + wave * 16;
    const int n0 = blockIdx.y * 64;

    // staging assignment: thread -> (column, 16B segment of the 64B K-chunk)
    const int colL = tid >> 2;                                  // 0..63
    const int seg  = tid & 3;                                   // 0..3
    const size_t gOff = (size_t)(n0 + colL) * N4K + seg * 8;    // in halves
    const unsigned ldsColOff = (unsigned)(colL * COL_STRIDE + seg * 16);
    const unsigned ldsBase = (unsigned)(size_t)(&smem[0]);      // LDS aperture: low 32 bits

    v8f acc[4] = { {}, {}, {}, {} };
    const _Float16* arow = Ah + (size_t)(m0 + r) * N4K;

    // prologue: stage k0 = 0 into buffer 0
    {
        unsigned long long gh = (unsigned long long)(Xh + gOff);
        unsigned long long gl = (unsigned long long)(Xl + gOff);
        unsigned dh = ldsBase + ldsColOff;
        unsigned dl = dh + MAT_BYTES;
        asm volatile("global_load_async_to_lds_b128 %0, %1, off" :: "v"(dh), "v"(gh) : "memory");
        asm volatile("global_load_async_to_lds_b128 %0, %1, off" :: "v"(dl), "v"(gl) : "memory");
    }

    for (int it = 0; it < N4K / 32; ++it) {
        const int k0 = it * 32;
        // my async copies for the current buffer have landed in LDS…
        asm volatile("s_wait_asynccnt 0x0" ::: "memory");
        // …and after the barrier, everyone's have; previous buffer is free.
        __syncthreads();

        if (it + 1 < N4K / 32) {       // stage next step into the other buffer
            unsigned long long gh = (unsigned long long)(Xh + gOff + (k0 + 32));
            unsigned long long gl = (unsigned long long)(Xl + gOff + (k0 + 32));
            unsigned dh = ldsBase + (unsigned)(((it + 1) & 1) * BUF_BYTES) + ldsColOff;
            unsigned dl = dh + MAT_BYTES;
            asm volatile("global_load_async_to_lds_b128 %0, %1, off" :: "v"(dh), "v"(gh) : "memory");
            asm volatile("global_load_async_to_lds_b128 %0, %1, off" :: "v"(dl), "v"(gl) : "memory");
        }

        // A 16x32 f16 fragment from global (HBM-bound stream, full-line coalesced):
        // lanes 0-15 hold K{0..15}; lanes 16-31 hold K{16..31}
        union { v16h v; v8h h[2]; } a;
        a.h[0] = *(const v8h*)(arow + k0 + hh * 16);
        a.h[1] = *(const v8h*)(arow + k0 + hh * 16 + 8);

        const unsigned char* bufp = smem + (unsigned)((it & 1) * BUF_BYTES);
#pragma unroll
        for (int t = 0; t < 4; ++t) {
            // B fragment from LDS: col = t*16+r, lane half 0: K0..15, half 1: K16..31
            const unsigned off = (unsigned)((t * 16 + r) * COL_STRIDE + hh * 32);
            union { v16h v; v8h h[2]; } bh, bl;
            bh.h[0] = *(const v8h*)(bufp + off);
            bh.h[1] = *(const v8h*)(bufp + off + 16);
            bl.h[0] = *(const v8h*)(bufp + MAT_BYTES + off);
            bl.h[1] = *(const v8h*)(bufp + MAT_BYTES + off + 16);
            acc[t] = __builtin_amdgcn_wmma_f32_16x16x32_f16(
                false, a.v, false, bh.v, (short)0, acc[t], false, false);
            acc[t] = __builtin_amdgcn_wmma_f32_16x16x32_f16(
                false, a.v, false, bl.v, (short)0, acc[t], false, false);
        }
    }

#pragma unroll
    for (int t = 0; t < 4; ++t) {
        const int c = n0 + t * 16 + r;
#pragma unroll
        for (int i = 0; i < 8; ++i)   // C/D: VGPR i -> M = i + 8*laneHalf
            S[(size_t)(m0 + i + hh * 8) * NCOL + c] = acc[t][i];
    }
}

// ---------------- assemble y[8192,128] with identity-block terms ----------
__global__ void k_assy(const float* __restrict__ S, const float* __restrict__ x,
                       const float* __restrict__ dinv, float* __restrict__ y) {
    int idx = blockIdx.x * blockDim.x + threadIdx.x;   // 4096*128
    int i = idx >> 7, c = idx & 127;
    float di = dinv[i], di2 = di * di;
    y[(size_t)i * DD + c]         = di * S[(size_t)i * NCOL + c]      + di2 * x[(size_t)(N4K + i) * DD + c];
    y[(size_t)(N4K + i) * DD + c] = di * S[(size_t)i * NCOL + DD + c] + di2 * x[(size_t)i * DD + c];
}

// ---------------- GEMM2: h = y @ W (fp32 WMMA 16x16x4) + BN partial stats --
__global__ void __launch_bounds__(256) k_gemm2(const float* __restrict__ y,
                                               const float* __restrict__ W,
                                               float* __restrict__ h,
                                               float* __restrict__ gsum,
                                               float* __restrict__ gsq) {
    __shared__ float sh[32];
    const int tid  = threadIdx.x;
    const int lane = tid & 31, wave = tid >> 5;
    const int r = lane & 15, hh = lane >> 4;
    const int m0 = blockIdx.x * 128 + wave * 16;
    const int n0 = blockIdx.y * 16;
    if (tid < 32) sh[tid] = 0.0f;
    __syncthreads();

    v8f acc = {};
    const float* yrow = y + (size_t)(m0 + r) * DD;
#pragma unroll
    for (int k0 = 0; k0 < DD; k0 += 4) {
        v2f a = *(const v2f*)(yrow + k0 + hh * 2);       // A: lanes<16 K{0,1}, lanes>=16 K{2,3}
        v2f b;                                           // B: VGPR e -> K = k0+2*hh+e, col n0+r
        b.x = W[(size_t)(k0 + 2 * hh) * DD + n0 + r];
        b.y = W[(size_t)(k0 + 2 * hh + 1) * DD + n0 + r];
        acc = __builtin_amdgcn_wmma_f32_16x16x4_f32(
            false, a, false, b, (short)0, acc, false, false);
    }

    float s = 0.0f, q = 0.0f;
#pragma unroll
    for (int i = 0; i < 8; ++i) {
        float v = acc[i];
        h[(size_t)(m0 + i + hh * 8) * DD + n0 + r] = v;
        s += v; q += v * v;
    }
    s += __shfl_xor(s, 16, 32);
    q += __shfl_xor(q, 16, 32);
    if (lane < 16) { atomicAdd(&sh[r], s); atomicAdd(&sh[16 + r], q); }
    __syncthreads();
    if (tid < 16)       atomicAdd(&gsum[n0 + tid], sh[tid]);
    else if (tid < 32)  atomicAdd(&gsq[n0 + tid - 16], sh[tid]);
}

// ---------------- BN finalize + ReLU --------------------------------------
__global__ void k_bn(const float* __restrict__ h, const float* __restrict__ gsum,
                     const float* __restrict__ gsq, const float* __restrict__ gamma,
                     const float* __restrict__ beta, float* __restrict__ out) {
    __shared__ float sA[DD], sB[DD];
    int tid = threadIdx.x;
    if (tid < DD) {
        const float inv = 1.0f / (2.0f * N4K);
        float m = gsum[tid] * inv;
        float var = gsq[tid] * inv - m * m;
        float a = gamma[tid] * rsqrtf(var + 1e-5f);
        sA[tid] = a;
        sB[tid] = beta[tid] - m * a;
    }
    __syncthreads();
    size_t idx = (size_t)blockIdx.x * blockDim.x + tid;
    int c = (int)(idx & (DD - 1));
    float v = h[idx] * sA[c] + sB[c];
    out[idx] = v > 0.0f ? v : 0.0f;
}

// ---------------- host launcher -------------------------------------------
extern "C" void kernel_launch(void* const* d_in, const int* in_sizes, int n_in,
                              void* d_out, int out_size, void* d_ws, size_t ws_size,
                              hipStream_t stream) {
    const float* x     = (const float*)d_in[0];
    const float* aff   = (const float*)d_in[1];
    const float* W     = (const float*)d_in[2];
    const float* gamma = (const float*)d_in[3];
    const float* beta  = (const float*)d_in[4];

    char* ws = (char*)d_ws;
    _Float16* Ah  = (_Float16*)(ws);                 // 32 MB
    _Float16* Xh  = (_Float16*)(ws + 33554432ull);   // 2 MB
    _Float16* Xl  = (_Float16*)(ws + 35651584ull);   // 2 MB
    float*    S   = (float*)(ws + 37748736ull);      // 4 MB
    float*    y   = (float*)(ws + 41943040ull);      // 4 MB
    float*    h   = (float*)(ws + 46137344ull);      // 4 MB
    float*    dinv= (float*)(ws + 50331648ull);      // 16 KB
    float*    gsum= (float*)(ws + 50348032ull);      // 1 KB (sum|sumsq)
    float*    gsq = gsum + 128;
    float*    out = (float*)d_out;

    k_zero    <<<1, 256, 0, stream>>>(gsum);
    k_binarize<<<512, 256, 0, stream>>>(aff, Ah, dinv);
    k_build_xs<<<4096, 256, 0, stream>>>(x, dinv, Xh, Xl);
    k_gemm1   <<<dim3(32, 4), 256, 0, stream>>>(Ah, Xh, Xl, S);
    k_assy    <<<2048, 256, 0, stream>>>(S, x, dinv, y);
    k_gemm2   <<<dim3(64, 8), 256, 0, stream>>>(y, W, h, gsum, gsq);
    k_bn      <<<4096, 256, 0, stream>>>(h, gsum, gsq, gamma, beta, out);
}
